// VectorQuantizer_Instance_Vr_68685116998173
// MI455X (gfx1250) — matrix operand order repaired and checked by reading, not compiled, not run on hardware
//
#include <hip/hip_runtime.h>
#include <hip/hip_bf16.h>

// ---------------------------------------------------------------------------
// VQ-VAE forward for MI455X (gfx1250, wave32, WMMA).
//   inputs      : [8192, 1024] f32   (d_in[0])
//   embeddings  : [4096, 1024] f32   (d_in[1])
//   d_out (f32) : loss(1) | quantized_st(8388608) | perplexity(1) | encodings(33554432)
// Distance GEMM runs as compensated-bf16 WMMA (hi*hi + hi*lo + lo*hi, f32 acc)
// for ~fp32-accurate argmin at bf16 matrix-core rates.
// Workspace usage: ~50.4 MB.
// ---------------------------------------------------------------------------

#define VQ_B   8192
#define VQ_E   4096
#define VQ_DIM 1024
#define KPAIRS 512          // VQ_DIM/2 packed-bf16 dwords per row

typedef __attribute__((ext_vector_type(16))) __bf16 v16bf;
typedef __attribute__((ext_vector_type(8)))  float  v8f;

union FragU { unsigned int u[8]; v16bf v; };

static __device__ inline unsigned short f32_to_bf16(float f) {
    unsigned int u = __float_as_uint(f);
    unsigned int lsb = (u >> 16) & 1u;
    u += 0x7FFFu + lsb;                 // round-to-nearest-even
    return (unsigned short)(u >> 16);
}
static __device__ inline float bf16_to_f32(unsigned short h) {
    return __uint_as_float(((unsigned int)h) << 16);
}

static __device__ inline v8f wmma_bf16(v16bf a, v16bf b, v8f c) {
    // (neg_a, A, neg_b, B, c_mod, C, reuse_a, reuse_b)
    return __builtin_amdgcn_wmma_f32_16x16x32_bf16(false, a, false, b,
                                                   (short)0, c, false, false);
}

// 256-thread block sum reduction (result valid on thread 0 only)
static __device__ inline float blockReduce256(float v) {
    #pragma unroll
    for (int off = 16; off > 0; off >>= 1) v += __shfl_xor(v, off, 32);
    __shared__ float tmp[8];
    int lane = threadIdx.x & 31, w = threadIdx.x >> 5;
    if (lane == 0) tmp[w] = v;
    __syncthreads();
    float s = 0.0f;
    if (threadIdx.x == 0) {
        #pragma unroll
        for (int i = 0; i < 8; ++i) s += tmp[i];
    }
    return s;
}

// --------------------------- prep kernels ----------------------------------

// f32 -> (bf16 hi, bf16 lo) split, 4 elements/thread
__global__ __launch_bounds__(256) void vq_split_bf16(
    const float* __restrict__ src, unsigned short* __restrict__ hi,
    unsigned short* __restrict__ lo, int n4)
{
    int i = blockIdx.x * 256 + threadIdx.x;
    if (i >= n4) return;
    float4 f = ((const float4*)src)[i];
    ushort4 h, l;
    h.x = f32_to_bf16(f.x); l.x = f32_to_bf16(f.x - bf16_to_f32(h.x));
    h.y = f32_to_bf16(f.y); l.y = f32_to_bf16(f.y - bf16_to_f32(h.y));
    h.z = f32_to_bf16(f.z); l.z = f32_to_bf16(f.z - bf16_to_f32(h.z));
    h.w = f32_to_bf16(f.w); l.w = f32_to_bf16(f.w - bf16_to_f32(h.w));
    ((ushort4*)hi)[i] = h;
    ((ushort4*)lo)[i] = l;
}

// per-code ||e||^2 in f32; one block per code
__global__ __launch_bounds__(256) void vq_code_norm(
    const float* __restrict__ emb, float* __restrict__ codeNorm)
{
    int code = blockIdx.x;
    float4 e = ((const float4*)(emb + (size_t)code * VQ_DIM))[threadIdx.x];
    float s = e.x * e.x + e.y * e.y + e.z * e.z + e.w * e.w;
    float tot = blockReduce256(s);
    if (threadIdx.x == 0) codeNorm[code] = tot;
}

__global__ void vq_zero_small(int* __restrict__ count, float* __restrict__ lossSum)
{
    int i = blockIdx.x * 256 + threadIdx.x;
    if (i < VQ_E) count[i] = 0;
    if (i == 0) *lossSum = 0.0f;
}

// --------------------------- distance + argmin -----------------------------
// One block = 4 waves, one 16-row M-tile. Wave w handles N-tile groups
// w, w+4, ... ; each group = 4 consecutive 16-code N-tiles (4 WMMA chains).
__global__ __launch_bounds__(128) void vq_dist_argmin(
    const unsigned int* __restrict__ xh, const unsigned int* __restrict__ xl,
    const unsigned int* __restrict__ eh, const unsigned int* __restrict__ el,
    const float* __restrict__ codeNorm, int* __restrict__ idxOut)
{
    __shared__ float sVal[4][16];
    __shared__ int   sIdx[4][16];

    const int tid   = threadIdx.x;
    const int wave  = tid >> 5;
    const int lane  = tid & 31;
    const int laneH = lane >> 4;        // 0: lanes 0-15, 1: lanes 16-31
    const int laneR = lane & 15;
    const int rowBase = blockIdx.x * 16;

    // A-matrix: lanes 0-15 and 16-31 both hold rows M=0..15 (different K)
    const unsigned int* xhRow = xh + (size_t)(rowBase + laneR) * KPAIRS;
    const unsigned int* xlRow = xl + (size_t)(rowBase + laneR) * KPAIRS;

    float minVal[8];
    int   minIdx[8];
    #pragma unroll
    for (int r = 0; r < 8; ++r) { minVal[r] = 3.4e38f; minIdx[r] = 0x7fffffff; }

    for (int g = wave; g < 64; g += 4) {              // 64 groups of 4 N-tiles
        const int nBase = g * 64;
        v8f acc[4];
        #pragma unroll
        for (int t = 0; t < 4; ++t) acc[t] = (v8f){0,0,0,0,0,0,0,0};

        for (int kc = 0; kc < 32; ++kc) {             // K = 1024, 32/step
            const int pairBase = kc * 16;
            // A fragment: pairs {0..3}+4h then {8..11}+4h
            FragU ah, al;
            *(uint4*)&ah.u[0] = *(const uint4*)(xhRow + pairBase + 4 * laneH);
            *(uint4*)&ah.u[4] = *(const uint4*)(xhRow + pairBase + 4 * laneH + 8);
            *(uint4*)&al.u[0] = *(const uint4*)(xlRow + pairBase + 4 * laneH);
            *(uint4*)&al.u[4] = *(const uint4*)(xlRow + pairBase + 4 * laneH + 8);
            #pragma unroll
            for (int t = 0; t < 4; ++t) {
                const int code = nBase + t * 16 + laneR;   // B column = laneR
                const unsigned int* ehp = eh + (size_t)code * KPAIRS + pairBase + 4 * laneH;
                const unsigned int* elp = el + (size_t)code * KPAIRS + pairBase + 4 * laneH;
                FragU bh, bl;
                *(uint4*)&bh.u[0] = *(const uint4*)(ehp);
                *(uint4*)&bh.u[4] = *(const uint4*)(ehp + 8);
                *(uint4*)&bl.u[0] = *(const uint4*)(elp);
                *(uint4*)&bl.u[4] = *(const uint4*)(elp + 8);
                acc[t] = wmma_bf16(ah.v, bh.v, acc[t]);   // hi*hi
                acc[t] = wmma_bf16(ah.v, bl.v, acc[t]);   // hi*lo
                acc[t] = wmma_bf16(al.v, bh.v, acc[t]);   // lo*hi
            }
        }
        // fold this group's 4 C-tiles into the running per-row argmin.
        // C layout: VGPR r -> row (r + 8*laneH), col laneR.
        #pragma unroll
        for (int t = 0; t < 4; ++t) {
            const int n  = nBase + t * 16 + laneR;
            const float cn = codeNorm[n];
            #pragma unroll
            for (int r = 0; r < 8; ++r) {
                float d = cn - 2.0f * acc[t][r];
                if (d < minVal[r] || (d == minVal[r] && n < minIdx[r])) {
                    minVal[r] = d; minIdx[r] = n;
                }
            }
        }
    }

    // reduce across the 16 columns within each lane-half
    #pragma unroll
    for (int off = 1; off < 16; off <<= 1) {
        #pragma unroll
        for (int r = 0; r < 8; ++r) {
            float ov = __shfl_xor(minVal[r], off, 32);
            int   oi = __shfl_xor(minIdx[r], off, 32);
            if (ov < minVal[r] || (ov == minVal[r] && oi < minIdx[r])) {
                minVal[r] = ov; minIdx[r] = oi;
            }
        }
    }
    if (laneR == 0) {
        #pragma unroll
        for (int r = 0; r < 8; ++r) {
            int row = r + 8 * laneH;
            sVal[wave][row] = minVal[r];
            sIdx[wave][row] = minIdx[r];
        }
    }
    __syncthreads();
    if (tid < 16) {
        float bv = sVal[0][tid]; int bi = sIdx[0][tid];
        #pragma unroll
        for (int w = 1; w < 4; ++w) {
            float v = sVal[w][tid]; int i = sIdx[w][tid];
            if (v < bv || (v == bv && i < bi)) { bv = v; bi = i; }
        }
        idxOut[rowBase + tid] = bi;
    }
}

// --------------------------- output kernels --------------------------------

// gather codebook row, straight-through output, loss partial. One block/row.
__global__ __launch_bounds__(256) void vq_gather(
    const float* __restrict__ x, const float* __restrict__ emb,
    const int* __restrict__ idx, float* __restrict__ quant,
    float* __restrict__ lossSum)
{
    const int b = blockIdx.x;
    const int code = idx[b];
    float4 e  = ((const float4*)(emb + (size_t)code * VQ_DIM))[threadIdx.x];
    float4 xv = ((const float4*)(x   + (size_t)b    * VQ_DIM))[threadIdx.x];
    float dx = e.x - xv.x, dy = e.y - xv.y, dz = e.z - xv.z, dw = e.w - xv.w;
    float* q = quant + (size_t)b * VQ_DIM + 4 * threadIdx.x;  // 4B-aligned dest
    q[0] = xv.x + dx; q[1] = xv.y + dy; q[2] = xv.z + dz; q[3] = xv.w + dw;
    float s = dx * dx + dy * dy + dz * dz + dw * dw;
    float tot = blockReduce256(s);
    if (threadIdx.x == 0) atomicAdd(lossSum, tot);
}

// zero encodings with float2 stores (base only 8B aligned)
__global__ void vq_enc_zero(float2* __restrict__ enc, long long n2)
{
    long long i = (long long)blockIdx.x * blockDim.x + threadIdx.x;
    long long stride = (long long)gridDim.x * blockDim.x;
    float2 z; z.x = 0.0f; z.y = 0.0f;
    for (; i < n2; i += stride) enc[i] = z;
}

__global__ void vq_onehot(float* __restrict__ enc, const int* __restrict__ idx,
                          int* __restrict__ count)
{
    int b = blockIdx.x * 256 + threadIdx.x;
    if (b < VQ_B) {
        int code = idx[b];
        enc[(size_t)b * VQ_E + code] = 1.0f;
        atomicAdd(&count[code], 1);
    }
}

__global__ __launch_bounds__(256) void vq_finalize(
    const int* __restrict__ count, const float* __restrict__ lossSum,
    float* __restrict__ outLoss, float* __restrict__ outPerp)
{
    float s = 0.0f;
    for (int i = threadIdx.x; i < VQ_E; i += 256) {
        float p = (float)count[i] * (1.0f / (float)VQ_B);
        s += p * logf(p + 1e-10f);
    }
    float tot = blockReduce256(s);
    if (threadIdx.x == 0) {
        *outPerp = expf(-tot);
        *outLoss = 1.25f * (*lossSum) / (float)((long long)VQ_B * VQ_DIM);
    }
}

// ------------------------------- launcher ----------------------------------

extern "C" void kernel_launch(void* const* d_in, const int* in_sizes, int n_in,
                              void* d_out, int out_size, void* d_ws, size_t ws_size,
                              hipStream_t stream)
{
    (void)in_sizes; (void)n_in; (void)out_size; (void)ws_size;
    const float* x   = (const float*)d_in[0];   // [8192,1024]
    const float* emb = (const float*)d_in[1];   // [4096,1024]
    float* out = (float*)d_out;

    // d_out layout
    float* outLoss = out;                         // [1]
    float* quant   = out + 1;                     // [8388608]
    float* outPerp = out + 1 + (size_t)VQ_B * VQ_DIM;          // [1]
    float* enc     = out + 2 + (size_t)VQ_B * VQ_DIM;          // [33554432]

    // workspace layout (~50.4 MB)
    char* w = (char*)d_ws;
    unsigned short* e_hi = (unsigned short*)(w);
    unsigned short* e_lo = (unsigned short*)(w + 8388608);
    unsigned short* x_hi = (unsigned short*)(w + 16777216);
    unsigned short* x_lo = (unsigned short*)(w + 33554432);
    float* codeNorm = (float*)(w + 50331648);
    int*   idx      = (int*)  (w + 50331648 + 16384);
    int*   count    = (int*)  (w + 50331648 + 16384 + 32768);
    float* lossSum  = (float*)(w + 50331648 + 16384 + 32768 + 16384);

    // 1. zero accumulators
    vq_zero_small<<<16, 256, 0, stream>>>(count, lossSum);
    // 2. compensated bf16 splits
    vq_split_bf16<<<(VQ_E * VQ_DIM / 4 + 255) / 256, 256, 0, stream>>>(emb, e_hi, e_lo, VQ_E * VQ_DIM / 4);
    vq_split_bf16<<<(VQ_B * VQ_DIM / 4 + 255) / 256, 256, 0, stream>>>(x, x_hi, x_lo, VQ_B * VQ_DIM / 4);
    // 3. code norms
    vq_code_norm<<<VQ_E, 256, 0, stream>>>(emb, codeNorm);
    // 4. WMMA distance GEMM + argmin
    vq_dist_argmin<<<VQ_B / 16, 128, 0, stream>>>(
        (const unsigned int*)x_hi, (const unsigned int*)x_lo,
        (const unsigned int*)e_hi, (const unsigned int*)e_lo, codeNorm, idx);
    // 5. gather + straight-through + loss partials
    vq_gather<<<VQ_B, 256, 0, stream>>>(x, emb, idx, quant, lossSum);
    // 6. encodings: zero then one-hot (+ histogram)
    vq_enc_zero<<<8192, 256, 0, stream>>>((float2*)enc, (long long)VQ_B * VQ_E / 2);
    vq_onehot<<<VQ_B / 256, 256, 0, stream>>>(enc, idx, count);
    // 7. scalars
    vq_finalize<<<1, 256, 0, stream>>>(count, lossSum, outLoss, outPerp);
}
